// LlamaAttention_6133213299167
// MI455X (gfx1250) — matrix operand order repaired and checked by reading, not compiled, not run on hardware
//
#include <hip/hip_runtime.h>
#include <hip/hip_bf16.h>
#include <math.h>

#define S_LEN 2048
#define DIM   2048
#define NH    32
#define DH    64

// GEMM block tiling
#define BM 128
#define BN 128
#define BK 32
#define LDP 40   // padded LDS row stride in halves (80B: 16B-aligned, bank-conflict-free)

typedef __attribute__((ext_vector_type(16))) _Float16 v16h;
typedef __attribute__((ext_vector_type(8)))  _Float16 v8h;
typedef __attribute__((ext_vector_type(8)))  float    v8f;
typedef int v4i_vec __attribute__((vector_size(16)));

#define AS1 __attribute__((address_space(1)))
#define AS3 __attribute__((address_space(3)))

#if __has_builtin(__builtin_amdgcn_global_load_async_to_lds_b128) && \
    __has_builtin(__builtin_amdgcn_s_wait_asynccnt)
#define HAVE_ASYNC_LDS 1
#else
#define HAVE_ASYNC_LDS 0
#endif

// ---------------------------------------------------------------------------
// WMMA fragment loaders following CDNA5 ISA VGPR layouts (CDNA5 ISA §7.12.2)
// A-matrix 16x32 f16: lane L (L<16): row M=L, K in {0..7, 16..23}
//                     lane L+16   : row M=L, K in {8..15, 24..31}
// ---------------------------------------------------------------------------
__device__ __forceinline__ v16h frag_ld_a(const _Float16* base, int stride, int lane) {
    const int r    = lane & 15;
    const int half = lane >> 4;
    const _Float16* p = base + (size_t)r * stride + half * 8;
    v8h lo = *(const v8h*)(p);        // K = half*8 .. half*8+7
    v8h hi = *(const v8h*)(p + 16);   // K = half*8+16 .. half*8+23
    v16h o;
#pragma unroll
    for (int i = 0; i < 8; ++i) { o[i] = lo[i]; o[8 + i] = hi[i]; }
    return o;
}

// B-matrix 32x16 f16: lane n (n<16): col N=n, K in {0..7, 16..23} ; lane n+16: K in {8..15, 24..31}
// (memory is B^T: row n of storage = column n, K contiguous)
__device__ __forceinline__ v16h frag_ld_b(const _Float16* base, int stride, int lane) {
    const int c    = lane & 15;
    const int half = lane >> 4;
    const _Float16* p = base + (size_t)c * stride + half * 8;
    v8h lo = *(const v8h*)(p);
    v8h hi = *(const v8h*)(p + 16);
    v16h o;
#pragma unroll
    for (int i = 0; i < 8; ++i) { o[i] = lo[i]; o[8 + i] = hi[i]; }
    return o;
}

__device__ __forceinline__ v8f wmma16(v16h a, v16h b, v8f c) {
    return __builtin_amdgcn_wmma_f32_16x16x32_f16(false, a, false, b,
                                                  (short)0, c, false, false);
}

// ---------------------------------------------------------------------------
// f32 -> f16 conversion
// ---------------------------------------------------------------------------
__global__ void cvt_f32_f16(const float* __restrict__ in, _Float16* __restrict__ out, int n) {
    int i = blockIdx.x * blockDim.x + threadIdx.x;
    if (i < n) out[i] = (_Float16)in[i];
}

// ---------------------------------------------------------------------------
// C[M,N] = A[M,K] * B^T (B stored N x K row-major).  f16 in, f32 out.
// 128x128 block tile, BK=32, double-buffered LDS (async-to-LDS when available),
// 8 waves; each wave computes 32x64 (2x4 WMMA accumulators).
// ---------------------------------------------------------------------------
__global__ __launch_bounds__(256)
void gemm_f16_nt(const _Float16* __restrict__ A,
                 const _Float16* __restrict__ B,
                 float* __restrict__ C, int M, int N, int K) {
    __shared__ _Float16 Asm[2][BM * LDP];
    __shared__ _Float16 Bsm[2][BN * LDP];

    const int tid  = threadIdx.x;
    const int lane = tid & 31;
    const int wave = tid >> 5;                 // 0..7
    const int mo   = blockIdx.y * BM;
    const int no   = blockIdx.x * BN;
    const int wm   = (wave >> 1) * 32;         // wave row offset in block tile
    const int wn   = (wave & 1) * 64;          // wave col offset in block tile

    // Tile staging: 128 rows x 32 halves = 512 b128 segments; 2 per thread.
    const int r0s = tid >> 1;
    const int k0s = (tid & 1) * 16;            // first 8-half segment
    const int k1s = k0s + 8;                   // second 8-half segment

    const int nk = K / BK;

    auto stage = [&](int buf, int kb) {
#if HAVE_ASYNC_LDS
        __builtin_amdgcn_global_load_async_to_lds_b128(
            (AS1 v4i_vec*)(A + (size_t)(mo + r0s) * K + kb + k0s),
            (AS3 v4i_vec*)(&Asm[buf][r0s * LDP + k0s]), 0, 0);
        __builtin_amdgcn_global_load_async_to_lds_b128(
            (AS1 v4i_vec*)(A + (size_t)(mo + r0s) * K + kb + k1s),
            (AS3 v4i_vec*)(&Asm[buf][r0s * LDP + k1s]), 0, 0);
        __builtin_amdgcn_global_load_async_to_lds_b128(
            (AS1 v4i_vec*)(B + (size_t)(no + r0s) * K + kb + k0s),
            (AS3 v4i_vec*)(&Bsm[buf][r0s * LDP + k0s]), 0, 0);
        __builtin_amdgcn_global_load_async_to_lds_b128(
            (AS1 v4i_vec*)(B + (size_t)(no + r0s) * K + kb + k1s),
            (AS3 v4i_vec*)(&Bsm[buf][r0s * LDP + k1s]), 0, 0);
#else
        v8h a0 = *(const v8h*)(A + (size_t)(mo + r0s) * K + kb + k0s);
        v8h a1 = *(const v8h*)(A + (size_t)(mo + r0s) * K + kb + k1s);
        v8h b0 = *(const v8h*)(B + (size_t)(no + r0s) * K + kb + k0s);
        v8h b1 = *(const v8h*)(B + (size_t)(no + r0s) * K + kb + k1s);
        *(v8h*)(&Asm[buf][r0s * LDP + k0s]) = a0;
        *(v8h*)(&Asm[buf][r0s * LDP + k1s]) = a1;
        *(v8h*)(&Bsm[buf][r0s * LDP + k0s]) = b0;
        *(v8h*)(&Bsm[buf][r0s * LDP + k1s]) = b1;
#endif
    };

    stage(0, 0);
#if HAVE_ASYNC_LDS
    __builtin_amdgcn_s_wait_asynccnt(0);
#endif
    __syncthreads();

    v8f acc[2][4] = {};
    for (int kt = 0; kt < nk; ++kt) {
        const int cur = kt & 1;
        if (kt + 1 < nk) stage(cur ^ 1, (kt + 1) * BK);

        v16h a0 = frag_ld_a(&Asm[cur][(wm +  0) * LDP], LDP, lane);
        v16h a1 = frag_ld_a(&Asm[cur][(wm + 16) * LDP], LDP, lane);
#pragma unroll
        for (int nt = 0; nt < 4; ++nt) {
            v16h b = frag_ld_b(&Bsm[cur][(wn + nt * 16) * LDP], LDP, lane);
            acc[0][nt] = wmma16(a0, b, acc[0][nt]);
            acc[1][nt] = wmma16(a1, b, acc[1][nt]);
        }
#if HAVE_ASYNC_LDS
        __builtin_amdgcn_s_wait_asynccnt(0);
#endif
        __syncthreads();
    }

    const int col = lane & 15;
    const int rr  = (lane >> 4) * 8;
#pragma unroll
    for (int mt = 0; mt < 2; ++mt)
#pragma unroll
        for (int nt = 0; nt < 4; ++nt)
#pragma unroll
            for (int i = 0; i < 8; ++i)
                C[(size_t)(mo + wm + mt * 16 + rr + i) * N +
                  (no + wn + nt * 16 + col)] = acc[mt][nt][i];
}

// ---------------------------------------------------------------------------
// RoPE for Q: qraw f32 (S,H,D) -> q16 f16 (S,H,D)
// ---------------------------------------------------------------------------
__global__ void rotary_q(const float* __restrict__ qraw, const float* __restrict__ fc,
                         const float* __restrict__ fs, _Float16* __restrict__ q16) {
    int idx = blockIdx.x * blockDim.x + threadIdx.x;       // over S*H*(D/2)
    if (idx >= S_LEN * NH * (DH / 2)) return;
    int p = idx & 31;
    int h = (idx >> 5) & 31;
    int s = idx >> 10;
    size_t base = (size_t)s * DIM + h * DH + 2 * p;
    float xr = qraw[base], xi = qraw[base + 1];
    float c = fc[s * (DH / 2) + p], sn = fs[s * (DH / 2) + p];
    q16[base]     = (_Float16)(xr * c - xi * sn);
    q16[base + 1] = (_Float16)(xr * sn + xi * c);
}

// RoPE + kv_mask + cache for K: writes final K f32 (d_out) and f16 copy.
// kraw may alias kout (elementwise read-before-write) -> no __restrict__.
__global__ void rotary_k(const float* kraw, const float* __restrict__ fc,
                         const float* __restrict__ fs, const float* __restrict__ kvm,
                         const float* __restrict__ kcache,
                         float* kout, _Float16* __restrict__ k16) {
    int idx = blockIdx.x * blockDim.x + threadIdx.x;
    if (idx >= S_LEN * NH * (DH / 2)) return;
    int p = idx & 31;
    int h = (idx >> 5) & 31;
    int s = idx >> 10;
    size_t base = (size_t)s * DIM + h * DH + 2 * p;
    float xr = kraw[base], xi = kraw[base + 1];
    float c = fc[s * (DH / 2) + p], sn = fs[s * (DH / 2) + p];
    float km = kvm[s];
    float vr = (xr * c - xi * sn) * km + kcache[base];
    float vi = (xr * sn + xi * c) * km + kcache[base + 1];
    kout[base]     = vr;
    kout[base + 1] = vi;
    k16[base]      = (_Float16)vr;
    k16[base + 1]  = (_Float16)vi;
}

// V: mask + cache, write final V f32 and a transposed f16 copy (H*D, S).
__global__ void v_update(const float* vraw, const float* __restrict__ kvm,
                         const float* __restrict__ vcache,
                         float* vout, _Float16* __restrict__ vT) {
    int idx = blockIdx.x * blockDim.x + threadIdx.x;       // over S*DIM
    if (idx >= S_LEN * DIM) return;
    int c = idx & (DIM - 1);
    int s = idx >> 11;
    float val = vraw[idx] * kvm[s] + vcache[idx];
    vout[idx] = val;
    vT[(size_t)c * S_LEN + s] = (_Float16)val;
}

// ---------------------------------------------------------------------------
// Flash-style attention: one wave per (16-query tile, head).
// Online softmax, QK^T and P*V via WMMA; P staged via LDS (C->A layout).
// ---------------------------------------------------------------------------
__global__ void attn_kernel(const _Float16* __restrict__ q16,
                            const _Float16* __restrict__ k16,
                            const _Float16* __restrict__ vT,
                            const float* __restrict__ mask,
                            _Float16* __restrict__ y16) {
    const int lane  = threadIdx.x & 31;
    const int qbase = blockIdx.x * 16;
    const int h     = blockIdx.y;
    const float scale = 0.125f;                       // 1/sqrt(64)
    __shared__ _Float16 Pl[16 * 32];

    v16h qa0 = frag_ld_a(q16 + (size_t)qbase * DIM + h * DH,      DIM, lane);
    v16h qa1 = frag_ld_a(q16 + (size_t)qbase * DIM + h * DH + 32, DIM, lane);

    const int col = lane & 15;
    const int r0  = (lane >> 4) * 8;                  // rows r0..r0+7 for this half

    float m[8], l[8];
    v8f acc[4] = {};
#pragma unroll
    for (int i = 0; i < 8; ++i) { m[i] = -INFINITY; l[i] = 0.f; }

    for (int kb = 0; kb < S_LEN; kb += 32) {
        float sv[2][8];
#pragma unroll
        for (int t = 0; t < 2; ++t) {
            v16h kb0 = frag_ld_b(k16 + (size_t)(kb + t * 16) * DIM + h * DH,      DIM, lane);
            v16h kb1 = frag_ld_b(k16 + (size_t)(kb + t * 16) * DIM + h * DH + 32, DIM, lane);
            v8f s = {};
            s = wmma16(qa0, kb0, s);
            s = wmma16(qa1, kb1, s);
#pragma unroll
            for (int i = 0; i < 8; ++i)
                sv[t][i] = s[i] * scale +
                    mask[(size_t)(qbase + r0 + i) * S_LEN + (kb + t * 16 + col)];
        }
        // per-row max across 32 columns (held by the 16 lanes of this half)
        float tm[8];
#pragma unroll
        for (int i = 0; i < 8; ++i) tm[i] = fmaxf(sv[0][i], sv[1][i]);
#pragma unroll
        for (int off = 1; off < 16; off <<= 1)
#pragma unroll
            for (int i = 0; i < 8; ++i)
                tm[i] = fmaxf(tm[i], __shfl_xor(tm[i], off, 32));

        float corr[8], rs[8];
#pragma unroll
        for (int i = 0; i < 8; ++i) {
            float mn = fmaxf(m[i], tm[i]);
            corr[i]  = __expf(m[i] - mn);
            float p0 = __expf(sv[0][i] - mn);
            float p1 = __expf(sv[1][i] - mn);
            Pl[(r0 + i) * 32 + col]      = (_Float16)p0;
            Pl[(r0 + i) * 32 + 16 + col] = (_Float16)p1;
            rs[i] = p0 + p1;
            m[i]  = mn;
        }
#pragma unroll
        for (int off = 1; off < 16; off <<= 1)
#pragma unroll
            for (int i = 0; i < 8; ++i) rs[i] += __shfl_xor(rs[i], off, 32);
#pragma unroll
        for (int i = 0; i < 8; ++i) l[i] = l[i] * corr[i] + rs[i];
#pragma unroll
        for (int t = 0; t < 4; ++t)
#pragma unroll
            for (int i = 0; i < 8; ++i) acc[t][i] *= corr[i];

        // P (16x32 f16, row-major in LDS) as A operand; V^T rows as B operand
        v16h pa = frag_ld_a(Pl, 32, lane);
#pragma unroll
        for (int t = 0; t < 4; ++t) {
            v16h vb = frag_ld_b(vT + (size_t)(h * DH + t * 16) * S_LEN + kb, S_LEN, lane);
            acc[t] = wmma16(pa, vb, acc[t]);
        }
    }
#pragma unroll
    for (int t = 0; t < 4; ++t)
#pragma unroll
        for (int i = 0; i < 8; ++i)
            y16[(size_t)(qbase + r0 + i) * DIM + h * DH + t * 16 + col] =
                (_Float16)(acc[t][i] / l[i]);
}

// ---------------------------------------------------------------------------
extern "C" void kernel_launch(void* const* d_in, const int* in_sizes, int n_in,
                              void* d_out, int out_size, void* d_ws, size_t ws_size,
                              hipStream_t stream) {
    const float* hidden = (const float*)d_in[0];
    const float* fcos   = (const float*)d_in[1];
    const float* fsin   = (const float*)d_in[2];
    const float* amask  = (const float*)d_in[3];
    const float* kvm    = (const float*)d_in[4];
    const float* kcache = (const float*)d_in[5];
    const float* vcache = (const float*)d_in[6];
    const float* wq     = (const float*)d_in[7];
    const float* wk     = (const float*)d_in[8];
    const float* wv     = (const float*)d_in[9];
    const float* wo     = (const float*)d_in[10];

    const size_t E = (size_t)S_LEN * DIM;             // 4,194,304 elements

    float* y_out = (float*)d_out;
    float* k_out = y_out + E;
    float* v_out = y_out + 2 * E;

    // d_out regions double as fp32 GEMM scratch (fully overwritten later,
    // elementwise-safe aliasing in rotary_k / v_update).
    float* qraw = y_out;
    float* kraw = k_out;
    float* vraw = v_out;

    char* ws = (char*)d_ws;
    _Float16* h16  = (_Float16*)ws; ws += E * 2;
    _Float16* wq16 = (_Float16*)ws; ws += E * 2;
    _Float16* wk16 = (_Float16*)ws; ws += E * 2;
    _Float16* wv16 = (_Float16*)ws; ws += E * 2;
    _Float16* wo16 = (_Float16*)ws; ws += E * 2;
    _Float16* q16  = (_Float16*)ws; ws += E * 2;
    _Float16* k16  = (_Float16*)ws; ws += E * 2;
    _Float16* vT16 = (_Float16*)ws; ws += E * 2;
    _Float16* y16  = (_Float16*)ws; ws += E * 2;

    const int CT = 256;
    const int cg = (int)((E + CT - 1) / CT);
    cvt_f32_f16<<<cg, CT, 0, stream>>>(hidden, h16,  (int)E);
    cvt_f32_f16<<<cg, CT, 0, stream>>>(wq,     wq16, (int)E);
    cvt_f32_f16<<<cg, CT, 0, stream>>>(wk,     wk16, (int)E);
    cvt_f32_f16<<<cg, CT, 0, stream>>>(wv,     wv16, (int)E);
    cvt_f32_f16<<<cg, CT, 0, stream>>>(wo,     wo16, (int)E);

    dim3 gg(DIM / BN, DIM / BM);
    gemm_f16_nt<<<gg, 256, 0, stream>>>(h16, wq16, qraw, DIM, DIM, DIM);
    gemm_f16_nt<<<gg, 256, 0, stream>>>(h16, wk16, kraw, DIM, DIM, DIM);
    gemm_f16_nt<<<gg, 256, 0, stream>>>(h16, wv16, vraw, DIM, DIM, DIM);

    const int RP = S_LEN * NH * (DH / 2);
    rotary_q<<<(RP + CT - 1) / CT, CT, 0, stream>>>(qraw, fcos, fsin, q16);
    rotary_k<<<(RP + CT - 1) / CT, CT, 0, stream>>>(kraw, fcos, fsin, kvm, kcache,
                                                    k_out, k16);
    v_update<<<cg, CT, 0, stream>>>(vraw, kvm, vcache, v_out, vT16);

    attn_kernel<<<dim3(S_LEN / 16, NH), 32, 0, stream>>>(q16, k16, vT16, amask, y16);

    gemm_f16_nt<<<gg, 256, 0, stream>>>(y16, wo16, y_out, DIM, DIM, DIM);
}
